// DGCNNAggregation_91156385890644
// MI455X (gfx1250) — matrix operand order repaired and checked by reading, not compiled, not run on hardware
//
#include <hip/hip_runtime.h>

typedef __attribute__((ext_vector_type(16))) _Float16 v16h;
typedef __attribute__((ext_vector_type(8)))  _Float16 v8h;
typedef __attribute__((ext_vector_type(8)))  float    v8f;

static constexpr int Bn = 8;
static constexpr int Nn = 2048;
static constexpr int Kn = 20;
static constexpr int NK = Nn * Kn;        // 40960
static constexpr int Pn = Bn * NK;        // 327680 positions total
static constexpr float EPSc  = 1e-5f;
static constexpr float SLOPE = 0.2f;

// ---------------------------------------------------------------------------
// Weight convert: f32 (cout,cin) -> f16 (cout,kpad) zero-padded
// ---------------------------------------------------------------------------
__global__ void convert_w_kernel(const float* __restrict__ W, _Float16* __restrict__ Wh,
                                 int cout, int cin, int kpad) {
  int i = blockIdx.x * 256 + threadIdx.x;
  if (i >= cout * kpad) return;
  int o = i / kpad, c = i % kpad;
  Wh[i] = (c < cin) ? (_Float16)W[o * cin + c] : (_Float16)0.0f;
}

// ---------------------------------------------------------------------------
// Squared norms xx[b,n] = sum_c x[b,c,n]^2
// ---------------------------------------------------------------------------
template<int C>
__global__ void sqnorm_kernel(const float* __restrict__ x, float* __restrict__ xx) {
  int t = blockIdx.x * 256 + threadIdx.x;          // b*N + n
  if (t >= Bn * Nn) return;
  int b = t / Nn, n = t % Nn;
  const float* xb = x + (size_t)b * C * Nn;
  float s = 0.f;
#pragma unroll
  for (int c = 0; c < C; ++c) { float v = xb[c * Nn + n]; s += v * v; }
  xx[t] = s;
}

// ---------------------------------------------------------------------------
// kNN top-K (largest pd = -||a-b||^2). One thread per query point, LDS tiles
// of candidate points broadcast to the block.
// ---------------------------------------------------------------------------
template<int C>
__global__ void __launch_bounds__(256) knn_kernel(const float* __restrict__ x,
                                                  const float* __restrict__ xx,
                                                  int* __restrict__ idx) {
  constexpr int TM = 32;
  __shared__ float tile[C * TM];
  __shared__ float txx[TM];
  int b = blockIdx.y;
  int n = blockIdx.x * 256 + threadIdx.x;
  const float* xb = x + (size_t)b * C * Nn;
  float ctr[C];
#pragma unroll
  for (int c = 0; c < C; ++c) ctr[c] = xb[c * Nn + n];
  float xn = xx[b * Nn + n];
  float bd[Kn]; int bi[Kn];
#pragma unroll
  for (int k = 0; k < Kn; ++k) { bd[k] = -3.4e38f; bi[k] = 0; }

  for (int m0 = 0; m0 < Nn; m0 += TM) {
    __syncthreads();
    for (int e = threadIdx.x; e < C * TM; e += 256) {
      int cc = e / TM, mm = e % TM;
      tile[e] = xb[cc * Nn + m0 + mm];
    }
    if (threadIdx.x < TM) txx[threadIdx.x] = xx[b * Nn + m0 + threadIdx.x];
    __syncthreads();
    for (int mm = 0; mm < TM; ++mm) {
      float dot = 0.f;
#pragma unroll
      for (int c = 0; c < C; ++c) dot += ctr[c] * tile[c * TM + mm];
      float d = 2.f * dot - xn - txx[mm];
      if (d > bd[Kn - 1]) {
        bd[Kn - 1] = d; bi[Kn - 1] = m0 + mm;
#pragma unroll
        for (int j = Kn - 1; j > 0; --j) {
          if (bd[j] > bd[j - 1]) {
            float tf = bd[j]; bd[j] = bd[j - 1]; bd[j - 1] = tf;
            int   ti = bi[j]; bi[j] = bi[j - 1]; bi[j - 1] = ti;
          }
        }
      }
    }
  }
  int* op = idx + (size_t)(b * Nn + n) * Kn;
#pragma unroll
  for (int k = 0; k < Kn; ++k) op[k] = bi[k];
}

// ---------------------------------------------------------------------------
// Edge features: feat[p, 0..C) = nbr-ctr, feat[p, C..2C) = ctr, pad -> 0. f16.
// Position-major (p, FD) so GEMM B-fragments are contiguous.
// ---------------------------------------------------------------------------
template<int C, int FD>
__global__ void feat_kernel(const float* __restrict__ x, const int* __restrict__ idx,
                            _Float16* __restrict__ feat) {
  int t = blockIdx.x * 256 + threadIdx.x;          // p = b*NK + n*K + k
  if (t >= Pn) return;
  int n = (t / Kn) % Nn;
  int b = t / NK;
  int j = idx[t];
  const float* xb = x + (size_t)b * C * Nn;
  _Float16* f = feat + (size_t)t * FD;
#pragma unroll
  for (int c = 0; c < C; ++c) {
    float ce = xb[c * Nn + n];
    float nb = xb[c * Nn + j];
    f[c]     = (_Float16)(nb - ce);
    f[C + c] = (_Float16)ce;
  }
#pragma unroll
  for (int c = 2 * C; c < FD; ++c) f[c] = (_Float16)0.0f;
}

// ---------------------------------------------------------------------------
// WMMA GEMM: y[p, o] = sum_i feat[p, i] * Wh[o, i].
// A = Wh tile (16x32 f16, from LDS), B = feat tile (32x16 f16), C = f32 16x16.
// 8 waves / block, each wave owns one 16-position tile and loops M-tiles.
// ---------------------------------------------------------------------------
template<int KPAD, int COUT>
__global__ void __launch_bounds__(256) gemm_kernel(const _Float16* __restrict__ Wh,
                                                   const _Float16* __restrict__ feat,
                                                   float* __restrict__ y) {
  __shared__ __align__(32) _Float16 wlds[COUT * KPAD];
  {
    const v8h* src = (const v8h*)Wh;
    v8h* dst = (v8h*)wlds;
    for (int i = threadIdx.x; i < COUT * KPAD / 8; i += 256) dst[i] = src[i];
  }
  __syncthreads();

  int wave = threadIdx.x >> 5;
  int lane = threadIdx.x & 31;
  int hi   = lane >> 4;            // half-wave select
  int ln   = lane & 15;
  int p    = (blockIdx.x * 8 + wave) * 16 + ln;    // this lane's output column

  constexpr int NKF = KPAD / 32;
  // B fragments: lane holds column p, 16 contiguous K-halfs (hi lanes K+16)
  v16h bf[NKF];
  const _Float16* fp = feat + (size_t)p * KPAD + hi * 16;
#pragma unroll
  for (int kf = 0; kf < NKF; ++kf) bf[kf] = *(const v16h*)(fp + kf * 32);

#pragma unroll
  for (int mt = 0; mt < COUT / 16; ++mt) {
    v8f acc = {};
    const _Float16* wp = &wlds[(mt * 16 + ln) * KPAD + hi * 16];
#pragma unroll
    for (int kf = 0; kf < NKF; ++kf) {
      v16h a = *(const v16h*)(wp + kf * 32);       // K {0..15} (+16 for hi lanes)
      acc = __builtin_amdgcn_wmma_f32_16x16x32_f16(false, a, false, bf[kf],
                                                   (short)0, acc, false, false);
    }
    float* yp = y + (size_t)p * COUT + mt * 16 + hi * 8;   // 8 contiguous floats
    *(v8f*)yp = acc;
  }
}

// ---------------------------------------------------------------------------
// BN: stats (sum, sumsq per channel), finalize scale/shift, apply, fused max
// ---------------------------------------------------------------------------
__global__ void zero_kernel(float* p, int n) {
  int i = blockIdx.x * 256 + threadIdx.x;
  if (i < n) p[i] = 0.f;
}

template<int COUT>
__global__ void bn_stats_kernel(const float* __restrict__ y, float* __restrict__ stats) {
  constexpr int G = 256 / COUT;
  constexpr int ITER = 64;
  constexpr int PPB = G * ITER;
  __shared__ float red[256];
  int o = threadIdx.x % COUT;
  int g = threadIdx.x / COUT;
  size_t p0 = (size_t)blockIdx.x * PPB + g;
  float s = 0.f, ss = 0.f;
  for (int i = 0; i < ITER; ++i) {
    float v = y[(p0 + (size_t)i * G) * COUT + o];
    s += v; ss += v * v;
  }
  red[threadIdx.x] = s;
  __syncthreads();
  if (g == 0) {
    float tot = 0.f;
#pragma unroll
    for (int gg = 0; gg < G; ++gg) tot += red[gg * COUT + o];
    atomicAdd(&stats[o], tot);
  }
  __syncthreads();
  red[threadIdx.x] = ss;
  __syncthreads();
  if (g == 0) {
    float tot = 0.f;
#pragma unroll
    for (int gg = 0; gg < G; ++gg) tot += red[gg * COUT + o];
    atomicAdd(&stats[COUT + o], tot);
  }
}

__global__ void bn_finalize_kernel(const float* __restrict__ stats,
                                   const float* __restrict__ gam,
                                   const float* __restrict__ bet,
                                   float* __restrict__ scsh, int cout) {
  int o = threadIdx.x;
  if (o >= cout) return;
  float cnt  = (float)Pn;
  float mean = stats[o] / cnt;
  float var  = stats[cout + o] / cnt - mean * mean;
  float inv  = rsqrtf(var + EPSc);
  float sc   = gam[o] * inv;
  scsh[o]        = sc;
  scsh[cout + o] = bet[o] - mean * sc;
}

template<int COUT>
__global__ void bn_apply_kernel(const float* __restrict__ y, const float* __restrict__ scsh,
                                _Float16* __restrict__ feat) {
  size_t i = (size_t)blockIdx.x * 256 + threadIdx.x;
  int o = (int)(i % COUT);
  float v = y[i] * scsh[o] + scsh[COUT + o];
  v = (v >= 0.f) ? v : SLOPE * v;
  feat[i] = (_Float16)v;
}

template<int COUT>
__global__ void bn_max_kernel(const float* __restrict__ y, const float* __restrict__ scsh,
                              float* __restrict__ out1, float* __restrict__ out2) {
  int t = blockIdx.x * 256 + threadIdx.x;          // b*COUT*N
  int n = t % Nn;
  int o = (t / Nn) % COUT;
  int b = t / (Nn * COUT);
  float sc = scsh[o], sh = scsh[COUT + o];
  size_t base = ((size_t)b * NK + (size_t)n * Kn) * COUT + o;
  float m = -3.4e38f;
#pragma unroll
  for (int k = 0; k < Kn; ++k) {
    float v = y[base + (size_t)k * COUT] * sc + sh;
    v = (v >= 0.f) ? v : SLOPE * v;
    m = fmaxf(m, v);
  }
  size_t oi = (size_t)b * COUT * Nn + (size_t)o * Nn + n;
  out1[oi] = m;
  if (out2) out2[oi] = m;
}

// ---------------------------------------------------------------------------
extern "C" void kernel_launch(void* const* d_in, const int* in_sizes, int n_in,
                              void* d_out, int out_size, void* d_ws, size_t ws_size,
                              hipStream_t stream) {
  (void)in_sizes; (void)n_in; (void)out_size; (void)ws_size;
  const float* x = (const float*)d_in[0];
  const float *W[5], *G[5], *Bp[5];
  for (int i = 0; i < 5; ++i) {
    W[i]  = (const float*)d_in[1 + 3 * i];
    G[i]  = (const float*)d_in[2 + 3 * i];
    Bp[i] = (const float*)d_in[3 + 3 * i];
  }

  // ---- workspace layout ----
  char* ws = (char*)d_ws;
  _Float16* Wh[5];
  for (int i = 0; i < 5; ++i) Wh[i] = (_Float16*)(ws + (size_t)i * 128 * 128 * 2);
  size_t off = (size_t)5 * 128 * 128 * 2;                        // 163840
  float* xx = (float*)(ws + off);      off += (size_t)Bn * Nn * 4;
  int*   idx = (int*)(ws + off);       off += (size_t)Bn * Nn * Kn * 4;
  _Float16* featA = (_Float16*)(ws + off); off += (size_t)Pn * 128 * 2;
  _Float16* featB = (_Float16*)(ws + off); off += (size_t)Pn * 128 * 2;
  float* ybuf  = (float*)(ws + off);   off += (size_t)Pn * 128 * 4;
  float* stats = (float*)(ws + off);   off += 256 * 4;
  float* scsh  = (float*)(ws + off);   off += 256 * 4;

  float* out  = (float*)d_out;
  float* xfin = out;                                  // (B,128,N)
  float* r1 = out + (size_t)Bn * 128 * Nn;            // (B,64,N)
  float* r2 = r1 + (size_t)Bn * 64 * Nn;              // (B,64,N)
  float* r3 = r2 + (size_t)Bn * 64 * Nn;              // (B,128,N)

  // ---- weights -> f16 padded ----
  const int cinv[5]  = {6, 64, 128, 64, 128};
  const int coutv[5] = {64, 64, 64, 64, 128};
  const int kpadv[5] = {32, 64, 128, 64, 128};
  for (int i = 0; i < 5; ++i) {
    int tot = coutv[i] * kpadv[i];
    convert_w_kernel<<<(tot + 255) / 256, 256, 0, stream>>>(W[i], Wh[i], coutv[i], cinv[i], kpadv[i]);
  }

  dim3 knngrid(Nn / 256, Bn);
  const int gemmgrid = Pn / (16 * 8);                 // 2560 blocks, 8 waves each

  // ======== Layer 1 (C=3, feat 6->32) ========
  sqnorm_kernel<3><<<(Bn * Nn) / 256, 256, 0, stream>>>(x, xx);
  knn_kernel<3><<<knngrid, 256, 0, stream>>>(x, xx, idx);
  feat_kernel<3, 32><<<Pn / 256, 256, 0, stream>>>(x, idx, featA);
  gemm_kernel<32, 64><<<gemmgrid, 256, 0, stream>>>(Wh[0], featA, ybuf);
  zero_kernel<<<1, 256, 0, stream>>>(stats, 256);
  bn_stats_kernel<64><<<Pn / 256, 256, 0, stream>>>(ybuf, stats);
  bn_finalize_kernel<<<1, 128, 0, stream>>>(stats, G[0], Bp[0], scsh, 64);
  bn_apply_kernel<64><<<(Pn / 256) * 64, 256, 0, stream>>>(ybuf, scsh, featB);
  gemm_kernel<64, 64><<<gemmgrid, 256, 0, stream>>>(Wh[1], featB, ybuf);
  zero_kernel<<<1, 256, 0, stream>>>(stats, 256);
  bn_stats_kernel<64><<<Pn / 256, 256, 0, stream>>>(ybuf, stats);
  bn_finalize_kernel<<<1, 128, 0, stream>>>(stats, G[1], Bp[1], scsh, 64);
  bn_max_kernel<64><<<Bn * 64 * Nn / 256, 256, 0, stream>>>(ybuf, scsh, r1, nullptr);

  // ======== Layer 2 (C=64, feat 128) ========
  sqnorm_kernel<64><<<(Bn * Nn) / 256, 256, 0, stream>>>(r1, xx);
  knn_kernel<64><<<knngrid, 256, 0, stream>>>(r1, xx, idx);
  feat_kernel<64, 128><<<Pn / 256, 256, 0, stream>>>(r1, idx, featA);
  gemm_kernel<128, 64><<<gemmgrid, 256, 0, stream>>>(Wh[2], featA, ybuf);
  zero_kernel<<<1, 256, 0, stream>>>(stats, 256);
  bn_stats_kernel<64><<<Pn / 256, 256, 0, stream>>>(ybuf, stats);
  bn_finalize_kernel<<<1, 128, 0, stream>>>(stats, G[2], Bp[2], scsh, 64);
  bn_apply_kernel<64><<<(Pn / 256) * 64, 256, 0, stream>>>(ybuf, scsh, featB);
  gemm_kernel<64, 64><<<gemmgrid, 256, 0, stream>>>(Wh[3], featB, ybuf);
  zero_kernel<<<1, 256, 0, stream>>>(stats, 256);
  bn_stats_kernel<64><<<Pn / 256, 256, 0, stream>>>(ybuf, stats);
  bn_finalize_kernel<<<1, 128, 0, stream>>>(stats, G[3], Bp[3], scsh, 64);
  bn_max_kernel<64><<<Bn * 64 * Nn / 256, 256, 0, stream>>>(ybuf, scsh, r2, nullptr);

  // ======== Layer 3 (C=64, feat 128, Cout=128) ========
  sqnorm_kernel<64><<<(Bn * Nn) / 256, 256, 0, stream>>>(r2, xx);
  knn_kernel<64><<<knngrid, 256, 0, stream>>>(r2, xx, idx);
  feat_kernel<64, 128><<<Pn / 256, 256, 0, stream>>>(r2, idx, featA);
  gemm_kernel<128, 128><<<gemmgrid, 256, 0, stream>>>(Wh[4], featA, ybuf);
  zero_kernel<<<1, 256, 0, stream>>>(stats, 256);
  bn_stats_kernel<128><<<Pn / 128, 256, 0, stream>>>(ybuf, stats);
  bn_finalize_kernel<<<1, 128, 0, stream>>>(stats, G[4], Bp[4], scsh, 128);
  bn_max_kernel<128><<<Bn * 128 * Nn / 256, 256, 0, stream>>>(ybuf, scsh, xfin, r3);
}
